// MultiHeadAttention_43456479101687
// MI455X (gfx1250) — matrix operand order repaired and checked by reading, compiled, not run on hardware
//
#include <hip/hip_runtime.h>

// ---------------------------------------------------------------------------
// Retention-style MHA block for MI455X (gfx1250), bf16 WMMA pipeline.
//   kqv = x@W1+b1 ; k,q,v split ; S = tril(q k^T) ; o = S v ; out = o@W2+b2
// v_wmma_f32_16x16x32_bf16 everywhere (f32 accumulate). Attention fused
// (512MB attn matrix never hits HBM). B/V staged transposed in LDS so all
// fragments load as ds_load_b128. A/K tiles staged via CDNA5 async
// global->LDS DMA (ASYNCcnt) into double buffers when available.
// ---------------------------------------------------------------------------

typedef __attribute__((ext_vector_type(16))) __bf16 v16bf;
typedef __attribute__((ext_vector_type(8)))  float  v8f;
typedef __attribute__((ext_vector_type(4)))  int    v4i;

#define D_EMBD  1024
#define N_HEADS 16
#define DH      64
#define BATCH   2
#define SEQ     2048
#define M_TOT   (BATCH * SEQ)   // 4096
#define N1      (3 * D_EMBD)    // 3072

#if __has_builtin(__builtin_amdgcn_global_load_async_to_lds_b128) && \
    __has_builtin(__builtin_amdgcn_s_wait_asynccnt)
#define ASYNC_LDS 1
#else
#define ASYNC_LDS 0
#endif

#define GAS __attribute__((address_space(1)))
#define LAS __attribute__((address_space(3)))

// 16B global->LDS copy: async DMA (ASYNCcnt) when available, sync fallback.
static __device__ __forceinline__ void cp_b128(const __bf16* g, __bf16* l) {
#if ASYNC_LDS
  __builtin_amdgcn_global_load_async_to_lds_b128(
      (GAS v4i*)(g), (LAS v4i*)(l), 0, 0);
#else
  const ulonglong2 u = *(const ulonglong2*)g;
  *(ulonglong2*)l = u;
#endif
}
template <int N>
static __device__ __forceinline__ void wait_async_c() {
#if ASYNC_LDS
  __builtin_amdgcn_s_wait_asynccnt((unsigned short)N);
#endif
}

static __device__ __forceinline__ __bf16 f2bf(float f) { return (__bf16)f; }
static __device__ __forceinline__ unsigned g16(unsigned long long w, int i) {
  return (unsigned)((w >> (16 * i)) & 0xFFFFu);
}

union AF { v16bf v; __bf16 e[16]; };
union CF { v8f   v; float  e[8];  };

// ---- WMMA fragment builders from LDS tiles (contiguous k order in A and B) --

// A-matrix 16x32 bf16: lanes 0-15 row M=lane, K=0..15; lanes 16-31 same rows,
// K=16..31 (contiguous, matches load_Bt_frag). Two 16B chunks -> 2x b128.
static __device__ __forceinline__ v16bf load_A_frag(const __bf16* base, int ld,
                                                    int mBase, int kBase, int lane) {
  const int half = lane >> 4;
  const int r    = lane & 15;
  const __bf16* row = base + (mBase + r) * ld + kBase;
  AF a;
#pragma unroll
  for (int v = 0; v < 4; ++v) {
    const int k = half * 16 + 2 * v;
    a.e[2 * v]     = row[k];
    a.e[2 * v + 1] = row[k + 1];
  }
#pragma unroll
  for (int v = 4; v < 8; ++v) {
    const int k = half * 16 + 2 * v;
    a.e[2 * v]     = row[k];
    a.e[2 * v + 1] = row[k + 1];
  }
  return a.v;
}

// B-matrix 32x16 bf16 where LDS holds BT[n][k] (column-major B): lane = column,
// lanes 0-15 K=0..15, lanes 16-31 K=16..31; contiguous per lane -> 2x b128.
static __device__ __forceinline__ v16bf load_Bt_frag(const __bf16* base, int ld,
                                                     int kBase, int nBase, int lane) {
  const int half = lane >> 4;
  const int n    = lane & 15;
  const __bf16* row = base + (nBase + n) * ld + kBase + half * 16;
  AF b;
#pragma unroll
  for (int v = 0; v < 8; ++v) {
    b.e[2 * v]     = row[2 * v];
    b.e[2 * v + 1] = row[2 * v + 1];
  }
  return b.v;
}

// ---------------------------- fp32 -> bf16 convert --------------------------
__global__ __launch_bounds__(256)
void cvt_bf16_kernel(const float4* __restrict__ in,
                     unsigned long long* __restrict__ out, int n4) {
  const int i = blockIdx.x * blockDim.x + threadIdx.x;
  if (i < n4) {
    const float4 f = in[i];
    union { __bf16 h[4]; unsigned long long u; } p;
    p.h[0] = f2bf(f.x); p.h[1] = f2bf(f.y);
    p.h[2] = f2bf(f.z); p.h[3] = f2bf(f.w);
    out[i] = p.u;
  }
}

// -------------------- bf16 GEMM: C = A[MxK] @ B[KxN] + bias -----------------
// Block tile 128x128, 8 waves (4x2), each wave 32x64 (2x4 WMMA tiles), K step 32.
// A staged via async DMA into double-buffered LDS; B staged transposed from
// pipelined registers (pack (k,k+1) pairs -> b32 stores).
template <bool OUT_BF16>
__global__ __launch_bounds__(256)
void gemm_bf16_kernel(const __bf16* __restrict__ A, const __bf16* __restrict__ B,
                      const float* __restrict__ bias,
                      __bf16* __restrict__ Cb, float* __restrict__ Cf,
                      int M, int N, int K) {
  __shared__ __align__(16) __bf16 As[2][128][40];  // double-buffered A tile
  __shared__ __align__(16) __bf16 BsT[128][40];    // transposed B tile

  const int tid  = threadIdx.x;
  const int lane = tid & 31;
  const int wid  = tid >> 5;
  const int wm   = wid >> 1;      // 0..3 -> 32-row strip
  const int wn   = wid & 1;       // 0..1 -> 64-col strip
  const int m0   = blockIdx.y * 128;
  const int n0   = blockIdx.x * 128;

  // staging maps
  const int arow = tid >> 1;            // 0..127
  const int ach  = (tid & 1) * 16;      // 0,16
  const int brow = (tid >> 4) * 2;      // 0..30 (pair of K rows)
  const int bnb  = (tid & 15) * 8;      // 0..120 (8 columns)

  const __bf16* Ag = A + (size_t)(m0 + arow) * K + ach;
  const __bf16* Bg = B + (size_t)brow * N + n0 + bnb;

  // prologue: A tile 0 -> buf0 (async), B tile 0 -> regs
  cp_b128(Ag,     &As[0][arow][ach]);
  cp_b128(Ag + 8, &As[0][arow][ach + 8]);
  ulonglong2 pb0 = *(const ulonglong2*)Bg;
  ulonglong2 pb1 = *(const ulonglong2*)(Bg + N);

  const v8f Z = {0.f, 0.f, 0.f, 0.f, 0.f, 0.f, 0.f, 0.f};
  v8f acc[2][4];
#pragma unroll
  for (int i = 0; i < 2; ++i)
#pragma unroll
    for (int j = 0; j < 4; ++j) acc[i][j] = Z;

  const int steps = K / 32;
  for (int it = 0; it < steps; ++it) {
    const int cur = it & 1;
    { // stage B tile 32x128 transposed: pack (k,k+1) pairs -> b32 stores
#pragma unroll
      for (int i = 0; i < 8; ++i) {
        const unsigned lo = g16(i < 4 ? pb0.x : pb0.y, i & 3);
        const unsigned hi = g16(i < 4 ? pb1.x : pb1.y, i & 3);
        *(unsigned*)&BsT[bnb + i][brow] = lo | (hi << 16);
      }
    }
    const bool hasNext = (it + 1 < steps);
    if (hasNext) { // async-fill the other A buffer; keep current tile complete
      Ag += 32;
      cp_b128(Ag,     &As[cur ^ 1][arow][ach]);
      cp_b128(Ag + 8, &As[cur ^ 1][arow][ach + 8]);
      wait_async_c<2>();
    } else {
      wait_async_c<0>();
    }
    __syncthreads();

    if (hasNext) { // prefetch next B tile into regs while WMMAs run
      Bg += (size_t)32 * N;
      pb0 = *(const ulonglong2*)Bg;
      pb1 = *(const ulonglong2*)(Bg + N);
    }

    const __bf16* Ab = &As[cur][0][0];
    const v16bf a0 = load_A_frag(Ab, 40, wm * 32,      0, lane);
    const v16bf a1 = load_A_frag(Ab, 40, wm * 32 + 16, 0, lane);
#pragma unroll
    for (int j = 0; j < 4; ++j) {
      const v16bf bb = load_Bt_frag(&BsT[0][0], 40, 0, wn * 64 + j * 16, lane);
      acc[0][j] = __builtin_amdgcn_wmma_f32_16x16x32_bf16(false, a0, false, bb, (short)0, acc[0][j], false, false);
      acc[1][j] = __builtin_amdgcn_wmma_f32_16x16x32_bf16(false, a1, false, bb, (short)0, acc[1][j], false, false);
    }
    __syncthreads();
  }

  // Epilogue: C layout = lane&15 -> column, VGPR v -> row v + 8*(lane>>4).
  const int half = lane >> 4;
  const int n    = lane & 15;
#pragma unroll
  for (int i = 0; i < 2; ++i) {
#pragma unroll
    for (int j = 0; j < 4; ++j) {
      const int col = n0 + wn * 64 + j * 16 + n;
      const float bv = bias[col];
      CF c; c.v = acc[i][j];
#pragma unroll
      for (int v = 0; v < 8; ++v) {
        const int row = m0 + wm * 32 + i * 16 + half * 8 + v;
        const float val = c.e[v] + bv;
        if constexpr (OUT_BF16) Cb[(size_t)row * N + col] = f2bf(val);
        else                    Cf[(size_t)row * N + col] = val;
      }
    }
  }
}

// ------------------- fused causal retention attention -----------------------
// One 128-thread WG per (b, h, 64 query rows). Key blocks (32 rows) streamed
// up to the diagonal: K via async DMA into double-buffered LDS, V transposed
// from pipelined regs. Mask only on diagonal blocks; fully-masked blocks skip
// compute per wave. S re-swizzled C->A layout via per-wave LDS scratch.
__global__ __launch_bounds__(128)
void attn_kernel(const __bf16* __restrict__ kqv, __bf16* __restrict__ attn_out) {
  __shared__ __align__(16) __bf16 Qs[64][72];       // Q strip, row-major
  __shared__ __align__(16) __bf16 Ks[2][32][72];    // K block x2, row-major
  __shared__ __align__(16) __bf16 VsT[64][40];      // V block, transposed
  __shared__ __align__(16) __bf16 Ss[4][16][40];    // per-wave S scratch

  const int tid   = threadIdx.x;
  const int lane  = tid & 31;
  const int w     = tid >> 5;          // wave 0..3 -> query rows w*16..w*16+15
  const int qb    = blockIdx.x;        // 0..31
  const int bh    = blockIdx.y;        // 0..31
  const int b     = bh >> 4;
  const int h     = bh & 15;
  const int qbase = qb * 64;
  const int qlo   = qbase + w * 16;    // wave's min query row
  const int qhi   = qlo + 15;          // wave's max query row

  // kqv row layout: [K(0..1023) | Q(1024..2047) | V(2048..3071)], head h at +h*64
  const __bf16* base = kqv + (size_t)b * SEQ * N1 + h * DH;

  { // stage Q strip 64x64: each thread 32 contiguous bf16 (64B)
    const int row = tid >> 1;
    const int ch  = (tid & 1) * 32;
    const __bf16* g = base + (size_t)(qbase + row) * N1 + D_EMBD + ch;
    const ulonglong2 u0 = *(const ulonglong2*)(g);
    const ulonglong2 u1 = *(const ulonglong2*)(g + 8);
    const ulonglong2 u2 = *(const ulonglong2*)(g + 16);
    const ulonglong2 u3 = *(const ulonglong2*)(g + 24);
    unsigned long long* l = (unsigned long long*)&Qs[row][ch];
    l[0] = u0.x; l[1] = u0.y; l[2] = u1.x; l[3] = u1.y;
    l[4] = u2.x; l[5] = u2.y; l[6] = u3.x; l[7] = u3.y;
  }
  __syncthreads();

  const v16bf aQ0 = load_A_frag(&Qs[0][0], 72, w * 16,  0, lane);
  const v16bf aQ1 = load_A_frag(&Qs[0][0], 72, w * 16, 32, lane);

  // staging maps: K row-major (async), V transposed pairs (regs)
  const int krow = tid >> 2;           // 0..31
  const int kqt  = (tid & 3) * 16;     // 0,16,32,48
  const int vrp  = (tid >> 3) * 2;     // 0..30 (pair of V rows)
  const int vnb  = (tid & 7) * 8;      // 0..56 (8 columns)

  auto issueK = [&](int kb, int buf) {
    const __bf16* gk = base + (size_t)(kb * 32 + krow) * N1 + kqt;
    cp_b128(gk,     &Ks[buf][krow][kqt]);
    cp_b128(gk + 8, &Ks[buf][krow][kqt + 8]);
  };
  ulonglong2 vx0, vx1;
  auto loadV = [&](int kb) {
    const __bf16* gv = base + (size_t)(kb * 32 + vrp) * N1 + 2 * D_EMBD + vnb;
    vx0 = *(const ulonglong2*)gv;
    vx1 = *(const ulonglong2*)(gv + N1);
  };
  issueK(0, 0);
  loadV(0);

  const v8f Z = {0.f, 0.f, 0.f, 0.f, 0.f, 0.f, 0.f, 0.f};
  v8f o0 = Z, o1 = Z, o2 = Z, o3 = Z;

  const int kbEnd = 2 * qb + 1;
  for (int kb = 0; kb <= kbEnd; ++kb) {
    const int cur = kb & 1;
    { // stage V block transposed: pack (k,k+1) pairs -> b32 stores
#pragma unroll
      for (int i = 0; i < 8; ++i) {
        const unsigned lo = g16(i < 4 ? vx0.x : vx0.y, i & 3);
        const unsigned hi = g16(i < 4 ? vx1.x : vx1.y, i & 3);
        *(unsigned*)&VsT[vnb + i][vrp] = lo | (hi << 16);
      }
    }
    const bool hasNext = (kb < kbEnd);
    if (hasNext) { issueK(kb + 1, cur ^ 1); wait_async_c<2>(); }
    else         { wait_async_c<0>(); }
    __syncthreads();

    if (hasNext) loadV(kb + 1);   // prefetch next V block while WMMAs run

    if (kb * 32 <= qhi) {         // else: block entirely above diagonal -> skip
      const bool needMask = (kb * 32 + 31) > qlo;
      const int half = lane >> 4;
      const int n    = lane & 15;
      const __bf16* Kb = &Ks[cur][0][0];
      // S(16x32) = Q(16x64) @ K^T(64x32), (masked,) -> bf16 in per-wave LDS
#pragma unroll
      for (int nt = 0; nt < 2; ++nt) {
        v8f s = Z;
        const v16bf bK0 = load_Bt_frag(Kb, 72,  0, nt * 16, lane);
        const v16bf bK1 = load_Bt_frag(Kb, 72, 32, nt * 16, lane);
        s = __builtin_amdgcn_wmma_f32_16x16x32_bf16(false, aQ0, false, bK0, (short)0, s, false, false);
        s = __builtin_amdgcn_wmma_f32_16x16x32_bf16(false, aQ1, false, bK1, (short)0, s, false, false);
        CF c; c.v = s;
        if (needMask) {
          const int kg = kb * 32 + nt * 16 + n;
#pragma unroll
          for (int v = 0; v < 8; ++v) {
            const int m  = half * 8 + v;
            const int qg = qlo + m;
            Ss[w][m][nt * 16 + n] = f2bf(kg <= qg ? c.e[v] : 0.0f);
          }
        } else {
#pragma unroll
          for (int v = 0; v < 8; ++v) {
            const int m = half * 8 + v;
            Ss[w][m][nt * 16 + n] = f2bf(c.e[v]);
          }
        }
      }

      // O(16x64) += S(16x32) @ V(32x64)
      const v16bf aS  = load_A_frag(&Ss[w][0][0], 40, 0, 0, lane);
      const v16bf bV0 = load_Bt_frag(&VsT[0][0], 40, 0,  0, lane);
      const v16bf bV1 = load_Bt_frag(&VsT[0][0], 40, 0, 16, lane);
      const v16bf bV2 = load_Bt_frag(&VsT[0][0], 40, 0, 32, lane);
      const v16bf bV3 = load_Bt_frag(&VsT[0][0], 40, 0, 48, lane);
      o0 = __builtin_amdgcn_wmma_f32_16x16x32_bf16(false, aS, false, bV0, (short)0, o0, false, false);
      o1 = __builtin_amdgcn_wmma_f32_16x16x32_bf16(false, aS, false, bV1, (short)0, o1, false, false);
      o2 = __builtin_amdgcn_wmma_f32_16x16x32_bf16(false, aS, false, bV2, (short)0, o2, false, false);
      o3 = __builtin_amdgcn_wmma_f32_16x16x32_bf16(false, aS, false, bV3, (short)0, o3, false, false);
    }
    __syncthreads();
  }

  // Merge heads: attn_out[b, t, h*64 + d] (bf16)
  const int half = lane >> 4;
  const int n    = lane & 15;
  v8f oo[4] = {o0, o1, o2, o3};
#pragma unroll
  for (int j = 0; j < 4; ++j) {
    CF c; c.v = oo[j];
#pragma unroll
    for (int v = 0; v < 8; ++v) {
      const int m = half * 8 + v;
      const int t = qbase + w * 16 + m;
      attn_out[((size_t)b * SEQ + t) * D_EMBD + h * DH + j * 16 + n] = f2bf(c.e[v]);
    }
  }
}

// ------------------------------- launcher -----------------------------------
extern "C" void kernel_launch(void* const* d_in, const int* in_sizes, int n_in,
                              void* d_out, int out_size, void* d_ws, size_t ws_size,
                              hipStream_t stream) {
  const float* x  = (const float*)d_in[0];
  const float* W1 = (const float*)d_in[1];
  const float* b1 = (const float*)d_in[2];
  const float* W2 = (const float*)d_in[3];
  const float* b2 = (const float*)d_in[4];
  float* out = (float*)d_out;

  // Workspace layout (bf16 elements): x | W1 | W2 | kqv | attn_out  (~48 MB)
  __bf16* xb   = (__bf16*)d_ws;
  __bf16* w1b  = xb   + (size_t)M_TOT * D_EMBD;   // 4,194,304
  __bf16* w2b  = w1b  + (size_t)D_EMBD * N1;      // 3,145,728
  __bf16* kqvb = w2b  + (size_t)D_EMBD * D_EMBD;  // 1,048,576
  __bf16* attb = kqvb + (size_t)M_TOT * N1;       // 12,582,912

  const int nx = M_TOT * D_EMBD / 4;
  cvt_bf16_kernel<<<(nx + 255) / 256, 256, 0, stream>>>(
      (const float4*)x, (unsigned long long*)xb, nx);
  const int nw1 = D_EMBD * N1 / 4;
  cvt_bf16_kernel<<<(nw1 + 255) / 256, 256, 0, stream>>>(
      (const float4*)W1, (unsigned long long*)w1b, nw1);
  const int nw2 = D_EMBD * D_EMBD / 4;
  cvt_bf16_kernel<<<(nw2 + 255) / 256, 256, 0, stream>>>(
      (const float4*)W2, (unsigned long long*)w2b, nw2);

  // kqv = x @ W1 + b1   (bf16 out)
  gemm_bf16_kernel<true><<<dim3(N1 / 128, M_TOT / 128), 256, 0, stream>>>(
      xb, w1b, b1, kqvb, nullptr, M_TOT, N1, D_EMBD);

  // fused masked retention attention -> merged heads (bf16)
  attn_kernel<<<dim3(SEQ / 64, BATCH * N_HEADS), 128, 0, stream>>>(kqvb, attb);

  // out = attn @ W2 + b2   (fp32 out)
  gemm_bf16_kernel<false><<<dim3(D_EMBD / 128, M_TOT / 128), 256, 0, stream>>>(
      attb, w2b, b2, nullptr, out, M_TOT, D_EMBD, D_EMBD);
}
